// MultiHeadAttention_61959198212419
// MI455X (gfx1250) — matrix operand order repaired and checked by reading, not compiled, not run on hardware
//
#include <hip/hip_runtime.h>
#include <hip/hip_bf16.h>

#define B_  2
#define T_  2048
#define D_  1024
#define H_  16
#define DH_ 64
#define NEG_INF_ (-1e30f)

typedef __attribute__((ext_vector_type(16))) __bf16        v16bf;
typedef __attribute__((ext_vector_type(8)))  float         v8f;

struct U8 { unsigned int v[8]; };

// fp32 -> bf16 round-to-nearest-even
static __device__ __forceinline__ unsigned short f2bf(float f) {
    unsigned int x = __float_as_uint(f);
    x += ((x >> 16) & 1u) + 0x7fffu;
    return (unsigned short)(x >> 16);
}
static __device__ __forceinline__ unsigned int pack2bf(float lo, float hi) {
    return (unsigned int)f2bf(lo) | ((unsigned int)f2bf(hi) << 16);
}

// --- CDNA5 async global->LDS copy (pure copy path, tracked by ASYNCcnt) ----
static __device__ __forceinline__ void async_copy_b128(void* lds_dst, const void* gsrc) {
    unsigned lds = (unsigned)(size_t)lds_dst;     // low 32 bits = LDS offset
    asm volatile("global_load_async_to_lds_b128 %0, %1, off"
                 :: "v"(lds), "v"(gsrc) : "memory");
}
static __device__ __forceinline__ void async_wait0() {
    asm volatile("s_wait_asynccnt 0" ::: "memory");
}

// A-fragment: 16x32 bf16, M = lane&15, lane-half selects K offset (0 / 16);
// regs 0..7 -> K 0..15(+off). LDS tile [row][k/2] dwords.
template <int P>
static __device__ __forceinline__ v16bf load_a_frag(const unsigned int (*buf)[P], int row0, int koff, int lane) {
    const int ml = lane & 15, hi = lane >> 4;
    U8 u;
#pragma unroll
    for (int j = 0; j < 8; ++j) {
        int idx = hi * 8 + j;
        u.v[j] = buf[row0 + ml][koff + idx];
    }
    return __builtin_bit_cast(v16bf, u);
}

// B-fragment: 32x16 bf16 (KxN). LDS holds B transposed: [n][k/2] dwords.
// lanes 0-15: N=lane, K=0..15; lanes 16-31: N=lane-16, K=16..31.
template <int P>
static __device__ __forceinline__ v16bf load_b_frag(const unsigned int (*buf)[P], int row0, int koff, int lane) {
    const int ml = lane & 15, hi = lane >> 4;
    U8 u;
#pragma unroll
    for (int j = 0; j < 8; ++j) u.v[j] = buf[row0 + ml][koff + hi * 8 + j];
    return __builtin_bit_cast(v16bf, u);
}

static __device__ __forceinline__ v8f wmma_bf16(v16bf a, v16bf b, v8f c) {
    return __builtin_amdgcn_wmma_f32_16x16x32_bf16(false, a, false, b, (short)0, c, false, false);
}

// ---------------------------------------------------------------------------
// C[M,N] = A[M,K] * W[N,K]^T + bias[N]
// mode 0: store bf16 split-head  [B,H,T,DH]
// mode 1: store fp32 row-major   [M,N]
// mode 2: store bf16 split-head transposed [B,H,DH,T]   (for V)
// Block tile 64x128, BK=32, 256 threads (8 waves, 2x4, each wave 32x32).
// fp32 inputs are converted to bf16 on the way into LDS (VALU path).
// ---------------------------------------------------------------------------
__global__ __launch_bounds__(256)
void gemm_bias_kernel(const float* __restrict__ A, const float* __restrict__ W,
                      const float* __restrict__ bias, void* __restrict__ dst,
                      int M, int N, int K, int mode)
{
    __shared__ unsigned int Abuf[64][17];
    __shared__ unsigned int Bbuf[128][17];
    const int tid  = threadIdx.x;
    const int lane = tid & 31, wave = tid >> 5;
    const int wm = wave >> 2, wn = wave & 3;
    const int blockM = blockIdx.y * 64, blockN = blockIdx.x * 128;

    const v8f z = {0.f, 0.f, 0.f, 0.f, 0.f, 0.f, 0.f, 0.f};
    v8f acc[2][2];
#pragma unroll
    for (int i = 0; i < 2; ++i)
#pragma unroll
        for (int j = 0; j < 2; ++j) acc[i][j] = z;

    for (int k0 = 0; k0 < K; k0 += 32) {
        const int c4 = tid & 7, rb = tid >> 3;
#pragma unroll
        for (int p = 0; p < 2; ++p) {   // A tile: 64 rows x 32 fp32 -> bf16
            int row = p * 32 + rb;
            const float4 v = *(const float4*)(A + (size_t)(blockM + row) * K + k0 + c4 * 4);
            Abuf[row][c4 * 2]     = pack2bf(v.x, v.y);
            Abuf[row][c4 * 2 + 1] = pack2bf(v.z, v.w);
        }
#pragma unroll
        for (int p = 0; p < 4; ++p) {   // B tile: 128 rows (n) x 32 fp32 (k) -> bf16
            int row = p * 32 + rb;
            const float4 v = *(const float4*)(W + (size_t)(blockN + row) * K + k0 + c4 * 4);
            Bbuf[row][c4 * 2]     = pack2bf(v.x, v.y);
            Bbuf[row][c4 * 2 + 1] = pack2bf(v.z, v.w);
        }
        __syncthreads();

        v16bf af[2], bf[2];
#pragma unroll
        for (int tm = 0; tm < 2; ++tm) af[tm] = load_a_frag(Abuf, wm * 32 + tm * 16, 0, lane);
#pragma unroll
        for (int tn = 0; tn < 2; ++tn) bf[tn] = load_b_frag(Bbuf, wn * 32 + tn * 16, 0, lane);
#pragma unroll
        for (int tm = 0; tm < 2; ++tm)
#pragma unroll
            for (int tn = 0; tn < 2; ++tn)
                acc[tm][tn] = wmma_bf16(af[tm], bf[tn], acc[tm][tn]);
        __syncthreads();
    }

    const int ml = lane & 15, hi = lane >> 4;
#pragma unroll
    for (int tm = 0; tm < 2; ++tm)
#pragma unroll
        for (int tn = 0; tn < 2; ++tn)
#pragma unroll
            for (int r = 0; r < 8; ++r) {
                int mg = blockM + wm * 32 + tm * 16 + hi * 8 + r;
                int ng = blockN + wn * 32 + tn * 16 + ml;
                float v = acc[tm][tn][r] + bias[ng];
                if (mode == 1) {
                    ((float*)dst)[(size_t)mg * N + ng] = v;
                } else {
                    int b = mg / T_, t = mg % T_;
                    int h = ng / DH_, dh = ng % DH_;
                    size_t idx = (mode == 0)
                        ? ((size_t)(b * H_ + h) * T_ + t) * DH_ + dh
                        : ((size_t)(b * H_ + h) * DH_ + dh) * T_ + t;
                    ((unsigned short*)dst)[idx] = f2bf(v);
                }
            }
}

// ---------------------------------------------------------------------------
// logits[bh,i,j] = scale * sum_d Q[bh,i,d]*K[bh,j,d]  (masked)
// Q,K bf16 [B*H, T, DH]. Block 64(q) x 128(k), 256 threads.
// Whole DH extent staged at once via async global->LDS b128 copies: one
// ASYNCcnt wait + one barrier, then both WMMA k-steps run from LDS.
// ---------------------------------------------------------------------------
__global__ __launch_bounds__(256)
void logits_kernel(const unsigned short* __restrict__ Qb, const unsigned short* __restrict__ Kb,
                   const unsigned char* __restrict__ amask, const int* __restrict__ causalp,
                   float* __restrict__ attn)
{
    __shared__ unsigned int Abuf[64][33];    // 64 q-rows x 32 dwords (DH=64 bf16)
    __shared__ unsigned int Bbuf[128][33];   // 128 k-rows x 32 dwords
    const int tid  = threadIdx.x;
    const int lane = tid & 31, wave = tid >> 5;
    const int wm = wave >> 2, wn = wave & 3;
    const int bh = blockIdx.z, b = bh / H_;
    const int blockM = blockIdx.y * 64;    // query rows
    const int blockN = blockIdx.x * 128;   // key cols
    const unsigned int* Qu = (const unsigned int*)Qb;
    const unsigned int* Ku = (const unsigned int*)Kb;

    // A tile: 64 rows x 8 b128-chunks = 512 chunks -> 2 per thread
#pragma unroll
    for (int p = 0; p < 2; ++p) {
        int chunk = p * 256 + tid;
        int row = chunk >> 3, c4 = chunk & 7;
        async_copy_b128(&Abuf[row][c4 * 4],
                        Qu + ((size_t)bh * T_ + blockM + row) * (DH_ / 2) + c4 * 4);
    }
    // B tile: 128 rows x 8 chunks = 1024 chunks -> 4 per thread
#pragma unroll
    for (int p = 0; p < 4; ++p) {
        int chunk = p * 256 + tid;
        int row = chunk >> 3, c4 = chunk & 7;
        async_copy_b128(&Bbuf[row][c4 * 4],
                        Ku + ((size_t)bh * T_ + blockN + row) * (DH_ / 2) + c4 * 4);
    }
    async_wait0();
    __syncthreads();

    const v8f z = {0.f, 0.f, 0.f, 0.f, 0.f, 0.f, 0.f, 0.f};
    v8f acc[2][2];
#pragma unroll
    for (int i = 0; i < 2; ++i)
#pragma unroll
        for (int j = 0; j < 2; ++j) acc[i][j] = z;

#pragma unroll
    for (int ks = 0; ks < 2; ++ks) {       // two 32-deep WMMA steps, koff in dwords
        const int koff = ks * 16;
        v16bf af[2], bf[2];
#pragma unroll
        for (int tm = 0; tm < 2; ++tm) af[tm] = load_a_frag(Abuf, wm * 32 + tm * 16, koff, lane);
#pragma unroll
        for (int tn = 0; tn < 2; ++tn) bf[tn] = load_b_frag(Bbuf, wn * 32 + tn * 16, koff, lane);
#pragma unroll
        for (int tm = 0; tm < 2; ++tm)
#pragma unroll
            for (int tn = 0; tn < 2; ++tn)
                acc[tm][tn] = wmma_bf16(af[tm], bf[tn], acc[tm][tn]);
    }

    const int causal = *causalp;
    const float scale = 0.125f;   // 1/sqrt(64)
    const int ml = lane & 15, hi = lane >> 4;
#pragma unroll
    for (int tm = 0; tm < 2; ++tm)
#pragma unroll
        for (int tn = 0; tn < 2; ++tn)
#pragma unroll
            for (int r = 0; r < 8; ++r) {
                int ig = blockM + wm * 32 + tm * 16 + hi * 8 + r;
                int jg = blockN + wn * 32 + tn * 16 + ml;
                bool keep = (amask[b * T_ + jg] != 0) && (!causal || jg <= ig);
                float v = keep ? acc[tm][tn][r] * scale : NEG_INF_;
                attn[((size_t)bh * T_ + ig) * T_ + jg] = v;
            }
}

// ---------------------------------------------------------------------------
// Row softmax in place over last dim (T_=2048). One 256-thread block per row.
// ---------------------------------------------------------------------------
__global__ __launch_bounds__(256)
void softmax_kernel(float* __restrict__ attn)
{
    const size_t row = blockIdx.x;
    float* p = attn + row * (size_t)T_;
    const int tid = threadIdx.x, lane = tid & 31, wave = tid >> 5;
    __shared__ float red[8];

    float x[8];
    float mx = -3.4e38f;
#pragma unroll
    for (int i = 0; i < 8; ++i) { x[i] = p[tid + i * 256]; mx = fmaxf(mx, x[i]); }
#pragma unroll
    for (int o = 16; o; o >>= 1) mx = fmaxf(mx, __shfl_xor(mx, o, 32));
    if (lane == 0) red[wave] = mx;
    __syncthreads();
    mx = red[0];
#pragma unroll
    for (int w = 1; w < 8; ++w) mx = fmaxf(mx, red[w]);
    __syncthreads();

    float s = 0.f;
#pragma unroll
    for (int i = 0; i < 8; ++i) { x[i] = __expf(x[i] - mx); s += x[i]; }
#pragma unroll
    for (int o = 16; o; o >>= 1) s += __shfl_xor(s, o, 32);
    if (lane == 0) red[wave] = s;
    __syncthreads();
    s = 0.f;
#pragma unroll
    for (int w = 0; w < 8; ++w) s += red[w];
    const float inv = 1.0f / s;
#pragma unroll
    for (int i = 0; i < 8; ++i) p[tid + i * 256] = x[i] * inv;
}

// ---------------------------------------------------------------------------
// ctx[bh,q,d] = sum_j attn[bh,q,j] * V[bh,j,d]
// attn fp32 (converted to bf16 into LDS on the VALU path), V held transposed
// bf16 [B*H, DH, T] and staged with async b128 copies. Block 64(q) x 64(d),
// 128 threads (4 waves 2x2, each 32x32). Causal bound on the K loop.
// ---------------------------------------------------------------------------
__global__ __launch_bounds__(128)
void av_kernel(const float* __restrict__ attn, const unsigned short* __restrict__ Vt,
               const int* __restrict__ causalp, float* __restrict__ ctx)
{
    __shared__ unsigned int Abuf[64][17];
    __shared__ unsigned int Bbuf[64][17];
    const int tid  = threadIdx.x;
    const int lane = tid & 31, wave = tid >> 5;
    const int wm = wave >> 1, wn = wave & 1;
    const int bh = blockIdx.y, b = bh / H_, h = bh % H_;
    const int blockM = blockIdx.x * 64;   // q rows
    const unsigned int* Vu = (const unsigned int*)Vt;

    const int causal = *causalp;
    const int kend = causal ? (blockM + 64) : T_;

    const v8f z = {0.f, 0.f, 0.f, 0.f, 0.f, 0.f, 0.f, 0.f};
    v8f acc[2][2];
#pragma unroll
    for (int i = 0; i < 2; ++i)
#pragma unroll
        for (int j = 0; j < 2; ++j) acc[i][j] = z;

    for (int k0 = 0; k0 < kend; k0 += 32) {
        {   // Vt tile: 64 d-rows x 4 b128-chunks = 256 chunks -> 2 per thread (async)
#pragma unroll
            for (int p = 0; p < 2; ++p) {
                int chunk = p * 128 + tid;
                int row = chunk >> 2, c4 = chunk & 3;
                async_copy_b128(&Bbuf[row][c4 * 4],
                                Vu + ((size_t)bh * DH_ + row) * (T_ / 2) + k0 / 2 + c4 * 4);
            }
        }
        {   // attn tile: 64 q-rows x 32 fp32 -> bf16 (VALU convert + ds_store)
            const int c4 = tid & 7, rb = tid >> 3;
#pragma unroll
            for (int p = 0; p < 4; ++p) {
                int row = p * 16 + rb;
                const float4 v = *(const float4*)(attn + ((size_t)bh * T_ + blockM + row) * T_ + k0 + c4 * 4);
                Abuf[row][c4 * 2]     = pack2bf(v.x, v.y);
                Abuf[row][c4 * 2 + 1] = pack2bf(v.z, v.w);
            }
        }
        async_wait0();
        __syncthreads();

        v16bf af[2], bf[2];
#pragma unroll
        for (int tm = 0; tm < 2; ++tm) af[tm] = load_a_frag(Abuf, wm * 32 + tm * 16, 0, lane);
#pragma unroll
        for (int tn = 0; tn < 2; ++tn) bf[tn] = load_b_frag(Bbuf, wn * 32 + tn * 16, 0, lane);
#pragma unroll
        for (int tm = 0; tm < 2; ++tm)
#pragma unroll
            for (int tn = 0; tn < 2; ++tn)
                acc[tm][tn] = wmma_bf16(af[tm], bf[tn], acc[tm][tn]);
        __syncthreads();
    }

    const int ml = lane & 15, hi = lane >> 4;
#pragma unroll
    for (int tm = 0; tm < 2; ++tm)
#pragma unroll
        for (int tn = 0; tn < 2; ++tn)
#pragma unroll
            for (int r = 0; r < 8; ++r) {
                int qg = blockM + wm * 32 + tm * 16 + hi * 8 + r;
                int dg = wn * 32 + tn * 16 + ml;             // 0..63
                ctx[((size_t)(b * T_ + qg)) * D_ + h * DH_ + dg] = acc[tm][tn][r];
            }
}

// ---------------------------------------------------------------------------
extern "C" void kernel_launch(void* const* d_in, const int* in_sizes, int n_in,
                              void* d_out, int out_size, void* d_ws, size_t ws_size,
                              hipStream_t stream)
{
    (void)in_sizes; (void)n_in; (void)out_size; (void)ws_size;

    const float*         query  = (const float*)d_in[0];
    const float*         key_   = (const float*)d_in[1];
    const float*         value  = (const float*)d_in[2];
    const unsigned char* amask  = (const unsigned char*)d_in[3];
    const int*           causal = (const int*)d_in[4];
    const float* Wq_w = (const float*)d_in[5];
    const float* Wq_b = (const float*)d_in[6];
    const float* Wk_w = (const float*)d_in[7];
    const float* Wk_b = (const float*)d_in[8];
    const float* Wv_w = (const float*)d_in[9];
    const float* Wv_b = (const float*)d_in[10];
    const float* Wo_w = (const float*)d_in[11];
    const float* Wo_b = (const float*)d_in[12];

    float* out  = (float*)d_out;
    float* attn = out + (size_t)B_ * T_ * D_;   // [B,H,T,T] region of d_out

    char* ws = (char*)d_ws;
    unsigned short* qb  = (unsigned short*)(ws);                            // bf16 [B,H,T,DH]  8 MB
    unsigned short* kb  = (unsigned short*)(ws + (size_t)8  * 1024 * 1024); // bf16 [B,H,T,DH]  8 MB
    unsigned short* vtb = (unsigned short*)(ws + (size_t)16 * 1024 * 1024); // bf16 [B,H,DH,T]  8 MB
    float*          ctx = (float*)        (ws + (size_t)24 * 1024 * 1024);  // fp32 [B,T,D]    16 MB

    const int M = B_ * T_;
    dim3 gProj(D_ / 128, M / 64);

    gemm_bias_kernel<<<gProj, 256, 0, stream>>>(query, Wq_w, Wq_b, qb,  M, D_, D_, 0);
    gemm_bias_kernel<<<gProj, 256, 0, stream>>>(key_,  Wk_w, Wk_b, kb,  M, D_, D_, 0);
    gemm_bias_kernel<<<gProj, 256, 0, stream>>>(value, Wv_w, Wv_b, vtb, M, D_, D_, 2);

    dim3 gLog(T_ / 128, T_ / 64, B_ * H_);
    logits_kernel<<<gLog, 256, 0, stream>>>(qb, kb, amask, causal, attn);

    softmax_kernel<<<(unsigned)(B_ * H_ * T_), 256, 0, stream>>>(attn);

    dim3 gAv(T_ / 64, B_ * H_);
    av_kernel<<<gAv, 128, 0, stream>>>(attn, vtb, causal, ctx);

    gemm_bias_kernel<<<gProj, 256, 0, stream>>>(ctx, Wo_w, Wo_b, out, M, D_, D_, 1);
}
